// GraphAttention_49684181680183
// MI455X (gfx1250) — compile-verified, matrix operand order checked by reading, not run on hardware
//
#include <hip/hip_runtime.h>

// ---------------------------------------------------------------------------
// Graph multi-head attention for MI455X (gfx1250, wave32, WMMA).
// B=8, N=2048, D=128, H=8, DH=16, O=128.
// Outputs (concatenated): output [B,N,O] f32, attn [B,H,N,N] f32.
// Roofline: attn write = 1.07 GB -> ~46us @ 23.3 TB/s; compute ~20 GFLOP (free).
// Kernel 2 is the hot kernel: per-lane online softmax (no per-tile cross-lane
// reductions), single butterfly merge per wave, attn written exactly once.
// Workspace: Qb/Kb (bf16 [B,H,N,16]), Vt (bf16 [B,H,16,N]), ctx (f32 [B,N,D])
//            = 20 MB total.
// ---------------------------------------------------------------------------

typedef __attribute__((ext_vector_type(16))) __bf16 v16bf;
typedef __attribute__((ext_vector_type(8)))  __bf16 v8bf;
typedef __attribute__((ext_vector_type(8)))  float  v8f;

#define BB 8
#define NN 2048
#define DD 128
#define HH 8
#define DHH 16
#define OO 128
#define NT (NN / 16)
#define NEGV (-1e9f)

// Native f32 -> bf16 (single v_cvt on gfx1250; bf16 VALU exists per ISA).
__device__ __forceinline__ __bf16 f2bf(float f) { return (__bf16)f; }
__device__ __forceinline__ unsigned short f2bfu(float f) {
    return __builtin_bit_cast(unsigned short, (__bf16)f);
}
__device__ __forceinline__ __bf16 bf0() {
    return __builtin_bit_cast(__bf16, (unsigned short)0);
}

// Build a K=32 bf16 WMMA operand from 32 consecutive-K f32 values.
// Per ISA 16-bit A/B layout: elements 0..7 = K(half)*8+{0..7}, 8..15 = +16.
__device__ __forceinline__ v16bf load_op_f32(const float* __restrict__ p) {
    v16bf r;
#pragma unroll
    for (int j = 0; j < 8; ++j) r[j] = f2bf(p[j]);
#pragma unroll
    for (int j = 0; j < 8; ++j) r[8 + j] = f2bf(p[16 + j]);
    return r;
}

// Build a K=16 (zero-padded to 32) operand from 8 consecutive bf16 (b128 load).
__device__ __forceinline__ v16bf load_op_bf16x8(const unsigned short* __restrict__ p) {
    v8bf lo = *(const v8bf*)p;
    v16bf r;
#pragma unroll
    for (int j = 0; j < 8; ++j) { r[j] = lo[j]; r[8 + j] = bf0(); }
    return r;
}

__device__ __forceinline__ v8f wmma_bf16(v16bf a, v16bf b, v8f c) {
    return __builtin_amdgcn_wmma_f32_16x16x32_bf16(
        /*neg_a=*/false, a, /*neg_b=*/false, b,
        /*c_mod=*/(short)0, c, /*reuse_a=*/false, /*reuse_b=*/false);
}

// xor-swizzle across lanes (group-of-32 mode): masks 1,2,4,8 stay inside the
// 16-lane halves, matching the WMMA C layout (row i in lanes 0-15 / 16-31).
template <int M>
__device__ __forceinline__ float swz(float v) {
    int r = __builtin_amdgcn_ds_swizzle(__builtin_bit_cast(int, v), (M << 10) | 0x1F);
    return __builtin_bit_cast(float, r);
}

// Merge two online-softmax partials (m = running max, l = running sum).
__device__ __forceinline__ void sm_merge(float& m, float& l, float om, float ol) {
    float nm = fmaxf(m, om);
    l = l * __expf(m - nm) + ol * __expf(om - nm);
    m = nm;
}

// ---------------------------------------------------------------------------
// Kernel 1: Q,K,V projections. 8 waves/block, wave w == head w (16 columns).
// ---------------------------------------------------------------------------
__global__ __launch_bounds__(256) void qkv_proj(
    const float* __restrict__ in, const float* __restrict__ Wq,
    const float* __restrict__ Wk, const float* __restrict__ Wv,
    unsigned short* __restrict__ Qb, unsigned short* __restrict__ Kb,
    unsigned short* __restrict__ Vt)
{
    const int wave = threadIdx.x >> 5;          // head / 16-col tile
    const int lane = threadIdx.x & 31;
    const int n  = lane & 15;
    const int hh = lane >> 4;
    const int b  = blockIdx.x / NT;
    const int qbase = (blockIdx.x % NT) * 16;

    const float* arow = in + (size_t)(b * NN + qbase + n) * DD + hh * 8;
    const float* wqr  = Wq + (size_t)(wave * 16 + n) * DD + hh * 8;
    const float* wkr  = Wk + (size_t)(wave * 16 + n) * DD + hh * 8;
    const float* wvr  = Wv + (size_t)(wave * 16 + n) * DD + hh * 8;

    v8f cq = {}, ck = {}, cv = {};
#pragma unroll
    for (int kc = 0; kc < DD; kc += 32) {
        v16bf a  = load_op_f32(arow + kc);
        v16bf bq = load_op_f32(wqr + kc);
        v16bf bk = load_op_f32(wkr + kc);
        v16bf bv = load_op_f32(wvr + kc);
        cq = wmma_bf16(a, bq, cq);
        ck = wmma_bf16(a, bk, ck);
        cv = wmma_bf16(a, bv, cv);
    }

    const size_t bh = (size_t)(b * HH + wave);
#pragma unroll
    for (int i = 0; i < 8; ++i) {
        const int row = i + hh * 8;                         // C layout: M=i(+8)
        Qb[(bh * NN + qbase + row) * DHH + n] = f2bfu(cq[i] * 0.25f); // 1/sqrt(16)
        Kb[(bh * NN + qbase + row) * DHH + n] = f2bfu(ck[i]);
        Vt[(bh * DHH + n) * NN + qbase + row] = f2bfu(cv[i]);         // transposed
    }
}

// ---------------------------------------------------------------------------
// Kernel 2: masked softmax attention. One wave per (b, h, 16-query tile).
// Sweep 1: per-lane online softmax stats (lane owns 128 columns of its row),
//          one butterfly merge at the end.
// Sweep 2: recompute scores, write attn once, ctx += P x V via second WMMA.
// ---------------------------------------------------------------------------
__global__ __launch_bounds__(32) void attn_kernel(
    const unsigned short* __restrict__ Qb, const unsigned short* __restrict__ Kb,
    const unsigned short* __restrict__ Vt, const unsigned char* __restrict__ mask,
    float* __restrict__ attn, float* __restrict__ ctx)
{
    __shared__ __bf16 pbuf[16 * 16];

    const int lane = threadIdx.x & 31;
    const int n  = lane & 15;
    const int hh = lane >> 4;
    const int b  = blockIdx.x / (HH * NT);
    const int h  = (blockIdx.x / NT) % HH;
    const int qbase = (blockIdx.x % NT) * 16;
    const size_t bh = (size_t)(b * HH + h);

    const unsigned short* Qp = Qb + bh * NN * DHH;
    const unsigned short* Kp = Kb + bh * NN * DHH;
    const unsigned short* Vp = Vt + bh * DHH * NN;

    // Q tile A-operand (K=16 zero-padded): lane row = n, K half = hh.
    const v16bf aq = load_op_bf16x8(Qp + (size_t)(qbase + n) * DHH + hh * 8);

    const unsigned char* mbase = mask + (size_t)(qbase + hh * 8) * NN + n;

    float mrun[8], lrun[8];
#pragma unroll
    for (int i = 0; i < 8; ++i) { mrun[i] = -3.0e38f; lrun[i] = 0.0f; }

    // ---- sweep 1: per-lane online max/sum (no cross-lane work in loop) ----
    for (int kt = 0; kt < NT; ++kt) {
        const int kbase = kt * 16;
        v16bf bk = load_op_bf16x8(Kp + (size_t)(kbase + n) * DHH + hh * 8);
        v8f c = {};
        c = wmma_bf16(aq, bk, c);
        const unsigned char* mr = mbase + kbase;
#pragma unroll
        for (int i = 0; i < 8; ++i) {
            float s  = mr[(size_t)i * NN] ? NEGV : c[i];
            float nm = fmaxf(mrun[i], s);
            lrun[i]  = lrun[i] * __expf(mrun[i] - nm) + __expf(s - nm);
            mrun[i]  = nm;
        }
    }
    // Butterfly merge across the 16-lane half (once per wave).
#pragma unroll
    for (int i = 0; i < 8; ++i) {
        float m = mrun[i], l = lrun[i];
        sm_merge(m, l, swz<1>(m), swz<1>(l));
        sm_merge(m, l, swz<2>(m), swz<2>(l));
        sm_merge(m, l, swz<4>(m), swz<4>(l));
        sm_merge(m, l, swz<8>(m), swz<8>(l));
        mrun[i] = m;
        lrun[i] = 1.0f / l;
    }

    // ---- sweep 2: write attn, accumulate ctx ----
    v8f cctx = {};
    for (int kt = 0; kt < NT; ++kt) {
        const int kbase = kt * 16;
        v16bf bk = load_op_bf16x8(Kp + (size_t)(kbase + n) * DHH + hh * 8);
        v8f c = {};
        c = wmma_bf16(aq, bk, c);
        const unsigned char* mr = mbase + kbase;
#pragma unroll
        for (int i = 0; i < 8; ++i) {
            const int row = i + hh * 8;
            float s = mr[(size_t)i * NN] ? NEGV : c[i];
            float p = __expf(s - mrun[i]) * lrun[i];
            attn[(bh * NN + qbase + row) * (size_t)NN + kbase + n] = p;
            pbuf[row * 16 + n] = f2bf(p);                 // stage for A-relayout
        }
        __syncthreads();
        // P as A-operand: lane row = n, K = hh*8 + j (ds_load_b128)
        v16bf ap;
        {
            const __bf16* pp = &pbuf[n * 16 + hh * 8];
#pragma unroll
            for (int j = 0; j < 8; ++j) { ap[j] = pp[j]; ap[8 + j] = bf0(); }
        }
        // V tile as B-operand: lane col = n (=dh), K = node kbase + hh*8 + j
        v16bf bv = load_op_bf16x8(Vp + (size_t)n * NN + kbase + hh * 8);
        cctx = wmma_bf16(ap, bv, cctx);
        __syncthreads();
    }

#pragma unroll
    for (int i = 0; i < 8; ++i) {
        const int row = i + hh * 8;
        ctx[(size_t)(b * NN + qbase + row) * DD + h * DHH + n] = cctx[i];
    }
}

// ---------------------------------------------------------------------------
// Kernel 3: output = input @ Wlq.T + ctx @ Wo.T.  8 waves/block, wave = o-tile.
// ---------------------------------------------------------------------------
__global__ __launch_bounds__(256) void out_proj(
    const float* __restrict__ in, const float* __restrict__ ctx,
    const float* __restrict__ Wo, const float* __restrict__ Wlq,
    float* __restrict__ out)
{
    const int wave = threadIdx.x >> 5;
    const int lane = threadIdx.x & 31;
    const int n  = lane & 15;
    const int hh = lane >> 4;
    const int b  = blockIdx.x / NT;
    const int qbase = (blockIdx.x % NT) * 16;

    const float* inr = in  + (size_t)(b * NN + qbase + n) * DD + hh * 8;
    const float* ctr = ctx + (size_t)(b * NN + qbase + n) * DD + hh * 8;
    const float* wlr = Wlq + (size_t)(wave * 16 + n) * DD + hh * 8;
    const float* wor = Wo  + (size_t)(wave * 16 + n) * DD + hh * 8;

    v8f acc = {};
#pragma unroll
    for (int kc = 0; kc < DD; kc += 32) {
        v16bf a1 = load_op_f32(inr + kc);
        v16bf b1 = load_op_f32(wlr + kc);
        acc = wmma_bf16(a1, b1, acc);
        v16bf a2 = load_op_f32(ctr + kc);
        v16bf b2 = load_op_f32(wor + kc);
        acc = wmma_bf16(a2, b2, acc);
    }
#pragma unroll
    for (int i = 0; i < 8; ++i) {
        const int row = i + hh * 8;
        out[(size_t)(b * NN + qbase + row) * OO + wave * 16 + n] = acc[i];
    }
}

// ---------------------------------------------------------------------------
extern "C" void kernel_launch(void* const* d_in, const int* in_sizes, int n_in,
                              void* d_out, int out_size, void* d_ws, size_t ws_size,
                              hipStream_t stream) {
    const float*         in   = (const float*)d_in[0];
    const unsigned char* mask = (const unsigned char*)d_in[1]; // jnp.bool_ = 1 byte
    const float*         Wq   = (const float*)d_in[2];
    const float*         Wk   = (const float*)d_in[3];
    const float*         Wv   = (const float*)d_in[4];
    const float*         Wo   = (const float*)d_in[5];
    const float*         Wlq  = (const float*)d_in[6];

    float* out  = (float*)d_out;
    float* attn = out + (size_t)BB * NN * OO;   // outputs concatenated in order

    const size_t nQKV = (size_t)BB * HH * NN * DHH;      // 2,097,152 elems
    unsigned short* Qb = (unsigned short*)d_ws;          //  4 MB
    unsigned short* Kb = Qb + nQKV;                      //  4 MB
    unsigned short* Vt = Kb + nQKV;                      //  4 MB
    float*          cx = (float*)(Vt + nQKV);            //  8 MB  (20 MB total)

    qkv_proj <<<BB * NT,      256, 0, stream>>>(in, Wq, Wk, Wv, Qb, Kb, Vt);
    attn_kernel<<<BB * HH * NT, 32, 0, stream>>>(Qb, Kb, Vt, mask, attn, cx);
    out_proj <<<BB * NT,      256, 0, stream>>>(in, cx, Wo, Wlq, out);
}
